// ModernAttention_29111288332610
// MI455X (gfx1250) — matrix operand order repaired and checked by reading, not compiled, not run on hardware
//
#include <hip/hip_runtime.h>
#include <hip/hip_bf16.h>
#include <math.h>

// ---------------------------------------------------------------------------
// Problem constants (from reference)
// ---------------------------------------------------------------------------
#define S_LEN  2048
#define BATCH  2
#define DMODEL 2048
#define NH     16
#define NKV    4
#define HD     128
#define NREP   (NH / NKV)

typedef __attribute__((ext_vector_type(16))) __bf16 bf16x16;
typedef __attribute__((ext_vector_type(8)))  float  f32x8;
typedef __attribute__((ext_vector_type(4)))  unsigned int u32x4;
typedef __attribute__((ext_vector_type(8)))  int    i32x8;
typedef __attribute__((ext_vector_type(4)))  int    i32x4;

#if __has_builtin(__builtin_amdgcn_tensor_load_to_lds) && \
    __has_builtin(__builtin_amdgcn_s_wait_tensorcnt)
#define USE_TDM 1
#else
#define USE_TDM 0
#endif

__device__ __forceinline__ unsigned short f32_to_bf16(float f) {
    union { float f; unsigned u; } x; x.f = f;
    unsigned r = x.u + 0x7FFFu + ((x.u >> 16) & 1u);   // round-to-nearest-even
    return (unsigned short)(r >> 16);
}

union ABFrag { uint4 u4[2]; bf16x16 v; };

#if USE_TDM
// Low 32 bits of a flat shared-memory address = LDS byte offset.
__device__ __forceinline__ unsigned lds_off(const void* p) {
    return (unsigned)(unsigned long long)(uintptr_t)p;
}

// Issue a TDM 2D tile load: tile_w bf16 elements per row, tile_h rows,
// row_stride elements between rows in global memory. Issued once per
// executing wave (EXEC ignored by tensor ops) -> call from one wave only.
__device__ __forceinline__ void tdm_load_2d(unsigned ldsaddr, const void* gptr,
                                            unsigned tile_w, unsigned tile_h,
                                            unsigned long long row_stride) {
    unsigned long long ga = (unsigned long long)(uintptr_t)gptr;
    u32x4 g0;
    g0[0] = 1u;                                   // count=1, user descriptor
    g0[1] = ldsaddr;                              // LDS byte address
    g0[2] = (unsigned)ga;                         // global_addr[31:0]
    g0[3] = (unsigned)((ga >> 32) & 0x01FFFFFFu)  // global_addr[56:32]
            | (2u << 30);                         // type = 2 ("image")
    const unsigned td0 = 0x40000000u;             // generous dims: no OOB clip
    const unsigned td1 = 0x00100000u;
    i32x8 g1;
    g1[0] = (int)(1u << 16);                      // data_size=1 (2 bytes/elem)
    g1[1] = (int)((td0 & 0xFFFFu) << 16);         // tensor_dim0[15:0]
    g1[2] = (int)((td0 >> 16) | ((td1 & 0xFFFFu) << 16));
    g1[3] = (int)((td1 >> 16) | (tile_w << 16));  // tile_dim0
    g1[4] = (int)(tile_h & 0xFFFFu);              // tile_dim1 (tile_dim2=0)
    g1[5] = (int)(unsigned)(row_stride & 0xFFFFFFFFull);   // dim0_stride[31:0]
    g1[6] = (int)(unsigned)((row_stride >> 32) & 0xFFFFull);
    g1[7] = 0;
    i32x4 gz = {0, 0, 0, 0};
#if __clang_major__ >= 23
    i32x8 gz8 = {0, 0, 0, 0, 0, 0, 0, 0};
    __builtin_amdgcn_tensor_load_to_lds(g0, g1, gz, gz, gz8, 0);
#else
    __builtin_amdgcn_tensor_load_to_lds(g0, g1, gz, gz, 0);
#endif
}
#endif // USE_TDM

// ---------------------------------------------------------------------------
// fp32 -> bf16 cast (grid-stride)
// ---------------------------------------------------------------------------
__global__ void cast_bf16_kernel(const float* __restrict__ src,
                                 unsigned short* __restrict__ dst, int n) {
    int i = blockIdx.x * blockDim.x + threadIdx.x;
    int stride = gridDim.x * blockDim.x;
    for (; i < n; i += stride) dst[i] = f32_to_bf16(src[i]);
}

// ---------------------------------------------------------------------------
// Tiled WMMA GEMM: C[M,N](f32) = A[M,K](bf16 rowmaj) * B[K,N](bf16 rowmaj)
// Block tile 128x128, K-step 32. 8 waves: 4 (M) x 2 (N); wave tile 32x64.
// A tile staged by the Tensor Data Mover (row-major copy); B tile transposed
// into column-major LDS by the threads.
// ---------------------------------------------------------------------------
__global__ __launch_bounds__(256)
void gemm_bf16_wmma(const unsigned short* __restrict__ A,
                    const unsigned short* __restrict__ B,
                    float* __restrict__ C, int M, int N, int K) {
    __shared__ unsigned short As[128 * 32];      // row-major [row][k]
    __shared__ unsigned short Bs[128 * 32];      // col-major [n][k]

    const int tid  = threadIdx.x;
    const int lane = tid & 31;
    const int wid  = tid >> 5;
    const int half = lane >> 4;
    const int l16  = lane & 15;
    const int waveM = wid & 3;                   // 0..3
    const int waveN = wid >> 2;                  // 0..1
    const int row0 = blockIdx.y * 128;
    const int col0 = blockIdx.x * 128;

    f32x8 acc[2][4];
#pragma unroll
    for (int a = 0; a < 2; a++)
#pragma unroll
        for (int b = 0; b < 4; b++) acc[a][b] = {};

    const int bk   = tid >> 3;                   // 0..31
    const int bn0  = (tid & 7) * 16;             // 0..112
#if !USE_TDM
    const int arow = tid >> 1;                   // 0..127
    const int acol = (tid & 1) * 16;             // 0 / 16
#endif

    for (int k0 = 0; k0 < K; k0 += 32) {
        // stage A tile: TDM row-major DMA (one wave issues, TENSORcnt-tracked)
#if USE_TDM
        if (wid == 0) {
            tdm_load_2d(lds_off(As), A + (size_t)row0 * K + k0,
                        /*tile_w=*/32, /*tile_h=*/128, /*stride=*/(unsigned long long)K);
        }
#else
        {
            const uint4* g = (const uint4*)(A + (size_t)(row0 + arow) * K + k0 + acol);
            uint4* s = (uint4*)(As + arow * 32 + acol);
            s[0] = g[0]; s[1] = g[1];
        }
#endif
        // stage B tile transposed into column-major LDS
        {
            union { uint4 u4[2]; unsigned short us[16]; } t;
            const unsigned short* gp = B + (size_t)(k0 + bk) * N + col0 + bn0;
            const uint4* g = (const uint4*)gp;
            t.u4[0] = g[0]; t.u4[1] = g[1];
            if (k0 + 32 < K)                      // prefetch next B tile
                __builtin_prefetch(gp + (size_t)32 * N, 0, 1);
#pragma unroll
            for (int i = 0; i < 16; i++) Bs[(bn0 + i) * 32 + bk] = t.us[i];
        }
#if USE_TDM
        if (wid == 0) __builtin_amdgcn_s_wait_tensorcnt(0);
#endif
        __syncthreads();

        ABFrag afrag[2];
#pragma unroll
        for (int mt = 0; mt < 2; mt++) {
            const unsigned short* p = As + (waveM * 32 + mt * 16 + l16) * 32 + half * 8;
            afrag[mt].u4[0] = *(const uint4*)(p);        // k = half*8 + 0..7
            afrag[mt].u4[1] = *(const uint4*)(p + 16);   // k = 16 + half*8 + 0..7
        }
#pragma unroll
        for (int nt = 0; nt < 4; nt++) {
            ABFrag bfrag;
            const unsigned short* p = Bs + (waveN * 64 + nt * 16 + l16) * 32 + half * 8;
            bfrag.u4[0] = *(const uint4*)(p);            // k = half*8 + 0..7
            bfrag.u4[1] = *(const uint4*)(p + 16);       // k = 16 + half*8 + 0..7
#pragma unroll
            for (int mt = 0; mt < 2; mt++) {
                acc[mt][nt] = __builtin_amdgcn_wmma_f32_16x16x32_bf16(
                    false, afrag[mt].v, false, bfrag.v, (short)0, acc[mt][nt],
                    false, false);
            }
        }
        __syncthreads();
    }

#pragma unroll
    for (int mt = 0; mt < 2; mt++)
#pragma unroll
        for (int nt = 0; nt < 4; nt++) {
            int c = col0 + waveN * 64 + nt * 16 + l16;
#pragma unroll
            for (int v = 0; v < 8; v++) {
                int r = row0 + waveM * 32 + mt * 16 + half * 8 + v;
                C[(size_t)r * N + c] = acc[mt][nt][v];
            }
        }
}

// ---------------------------------------------------------------------------
// RoPE + repack [S,B,nh,HD](f32) -> [B,nh,S,HD](bf16); scale folded in.
// ---------------------------------------------------------------------------
__global__ void rope_pack_kernel(const float* __restrict__ src,
                                 unsigned short* __restrict__ dst,
                                 int nheads, float scale) {
    int idx = blockIdx.x * blockDim.x + threadIdx.x;
    int total = S_LEN * BATCH * nheads * (HD / 2);
    if (idx >= total) return;
    int i2 = idx & 63;
    int h  = (idx >> 6) % nheads;
    int b  = (idx / (64 * nheads)) % BATCH;
    int s  = idx / (64 * nheads * BATCH);
    const float* q = src + ((size_t)(s * BATCH + b) * nheads + h) * HD;
    float q1 = q[i2], q2 = q[i2 + 64];
    float invf = __expf(-((float)i2 * (1.0f / 64.0f)) * 9.210340372f); // ln(1e4)
    float fr = (float)s * invf;
    float c, sn;
    __sincosf(fr, &sn, &c);
    float o1 = (q1 * c - q2 * sn) * scale;
    float o2 = (q2 * c + q1 * sn) * scale;
    size_t dbase = ((size_t)(b * nheads + h) * S_LEN + s) * HD;
    dst[dbase + i2]      = f32_to_bf16(o1);
    dst[dbase + i2 + 64] = f32_to_bf16(o2);
}

// ---------------------------------------------------------------------------
// V repack [S,B,NKV,HD](f32) -> [B,NKV,S,HD](bf16)
// ---------------------------------------------------------------------------
__global__ void pack_v_kernel(const float* __restrict__ src,
                              unsigned short* __restrict__ dst) {
    int idx = blockIdx.x * blockDim.x + threadIdx.x;
    int total = S_LEN * BATCH * NKV * HD;
    if (idx >= total) return;
    int d = idx & 127;
    int h = (idx >> 7) % NKV;
    int b = (idx / (128 * NKV)) % BATCH;
    int s = idx / (128 * NKV * BATCH);
    float v = src[((size_t)(s * BATCH + b) * NKV + h) * HD + d];
    dst[((size_t)(b * NKV + h) * S_LEN + s) * HD + d] = f32_to_bf16(v);
}

// ---------------------------------------------------------------------------
// Flash attention (causal, GQA). Grid: (S/128, NH, BATCH), 256 threads.
// K tile staged via TDM; V tile transposed by threads. Q pre-scaled.
// ---------------------------------------------------------------------------
__global__ __launch_bounds__(256)
void flash_attn_kernel(const unsigned short* __restrict__ Qb,
                       const unsigned short* __restrict__ Kb,
                       const unsigned short* __restrict__ Vb,
                       unsigned short* __restrict__ Ob) {
    __shared__ unsigned short Ks[64 * 128];      // [key][hd] row-major
    __shared__ unsigned short Vt[128 * 64];      // [hd][key] transposed
    __shared__ unsigned short Ps[8][16 * 64];    // per-wave P tile [m][key]

    const int tid  = threadIdx.x;
    const int lane = tid & 31;
    const int wid  = tid >> 5;
    const int half = lane >> 4;
    const int l16  = lane & 15;
    const int b    = blockIdx.z;
    const int h    = blockIdx.y;
    const int q0   = blockIdx.x * 128;
    const int kvh  = h / NREP;

    const unsigned short* Qbase = Qb + ((size_t)(b * NH  + h)   * S_LEN) * HD;
    const unsigned short* Kbase = Kb + ((size_t)(b * NKV + kvh) * S_LEN) * HD;
    const unsigned short* Vbase = Vb + ((size_t)(b * NKV + kvh) * S_LEN) * HD;

    const int qr0 = q0 + wid * 16;

    // Q fragments for 4 hd-chunks of 32 (A-matrix layout)
    ABFrag qf[4];
#pragma unroll
    for (int c = 0; c < 4; c++) {
        const unsigned short* p = Qbase + (size_t)(qr0 + l16) * HD + c * 32 + half * 8;
        qf[c].u4[0] = *(const uint4*)(p);
        qf[c].u4[1] = *(const uint4*)(p + 16);
    }

    f32x8 oacc[8];
#pragma unroll
    for (int j = 0; j < 8; j++) oacc[j] = {};
    float mi[8], li[8];
#pragma unroll
    for (int v = 0; v < 8; v++) { mi[v] = -3.0e38f; li[v] = 0.0f; }

    const int srow = wid * 16 + half * 8;
    const int wave_max_row = q0 + wid * 16 + 15;
    const int nkt = q0 / 64 + 2;

    const int krow = tid >> 2;                   // 0..63
    const int kcol = (tid & 3) * 32;             // 0..96

    for (int kt = 0; kt < nkt; kt++) {
        const int ktb = kt * 64;
        // stage K (row-major) via TDM
#if USE_TDM
        if (wid == 0) {
            tdm_load_2d(lds_off(Ks), Kbase + (size_t)ktb * HD,
                        /*tile_w=*/128, /*tile_h=*/64, /*stride=*/HD);
        }
#else
        {
            const uint4* g = (const uint4*)(Kbase + (size_t)(ktb + krow) * HD + kcol);
            uint4* s = (uint4*)(Ks + krow * 128 + kcol);
            s[0] = g[0]; s[1] = g[1]; s[2] = g[2]; s[3] = g[3];
        }
#endif
        // stage V transposed [hd][key]
        {
            union { uint4 u4[4]; unsigned short us[32]; } t;
            const uint4* g = (const uint4*)(Vbase + (size_t)(ktb + krow) * HD + kcol);
            t.u4[0] = g[0]; t.u4[1] = g[1]; t.u4[2] = g[2]; t.u4[3] = g[3];
#pragma unroll
            for (int i = 0; i < 32; i++) Vt[(kcol + i) * 64 + krow] = t.us[i];
        }
#if USE_TDM
        if (wid == 0) __builtin_amdgcn_s_wait_tensorcnt(0);
#endif
        __syncthreads();

        if (ktb <= wave_max_row) {               // wave-uniform: EXEC stays full
            // S = Q * K^T   (16 x 64 tile, fp32 accum)
            f32x8 sacc[4];
#pragma unroll
            for (int nt = 0; nt < 4; nt++) sacc[nt] = {};
#pragma unroll
            for (int nt = 0; nt < 4; nt++) {
#pragma unroll
                for (int c = 0; c < 4; c++) {
                    ABFrag bf;
                    const unsigned short* p = Ks + (nt * 16 + l16) * 128 + c * 32 + half * 8;
                    bf.u4[0] = *(const uint4*)(p);
                    bf.u4[1] = *(const uint4*)(p + 16);
                    sacc[nt] = __builtin_amdgcn_wmma_f32_16x16x32_bf16(
                        false, qf[c].v, false, bf.v, (short)0, sacc[nt], false, false);
                }
            }
            // causal mask
#pragma unroll
            for (int nt = 0; nt < 4; nt++) {
                int key = ktb + nt * 16 + l16;
#pragma unroll
                for (int v = 0; v < 8; v++) {
                    int row = q0 + srow + v;
                    if (key > row) sacc[nt][v] = -3.0e38f;
                }
            }
            // online softmax (row stats replicated across the 16-lane half)
            float alpha[8];
#pragma unroll
            for (int v = 0; v < 8; v++) {
                float m = fmaxf(fmaxf(sacc[0][v], sacc[1][v]),
                                fmaxf(sacc[2][v], sacc[3][v]));
#pragma unroll
                for (int msk = 1; msk < 16; msk <<= 1)
                    m = fmaxf(m, __shfl_xor(m, msk, 32));
                float mnew = fmaxf(mi[v], m);
                alpha[v] = __expf(mi[v] - mnew);
                float sum = 0.0f;
#pragma unroll
                for (int nt = 0; nt < 4; nt++) {
                    float p = __expf(sacc[nt][v] - mnew);
                    sacc[nt][v] = p;
                    sum += p;
                }
#pragma unroll
                for (int msk = 1; msk < 16; msk <<= 1)
                    sum += __shfl_xor(sum, msk, 32);
                li[v] = li[v] * alpha[v] + sum;
                mi[v] = mnew;
            }
#pragma unroll
            for (int j = 0; j < 8; j++)
#pragma unroll
                for (int v = 0; v < 8; v++) oacc[j][v] *= alpha[v];

            // C-layout P -> A-layout via per-wave LDS tile (bf16)
#pragma unroll
            for (int nt = 0; nt < 4; nt++)
#pragma unroll
                for (int v = 0; v < 8; v++)
                    Ps[wid][(half * 8 + v) * 64 + nt * 16 + l16] =
                        f32_to_bf16(sacc[nt][v]);

            // O += P * V
#pragma unroll
            for (int c2 = 0; c2 < 2; c2++) {
                ABFrag af;
                const unsigned short* p = &Ps[wid][l16 * 64 + c2 * 32 + half * 8];
                af.u4[0] = *(const uint4*)(p);
                af.u4[1] = *(const uint4*)(p + 16);
#pragma unroll
                for (int j = 0; j < 8; j++) {
                    ABFrag bf;
                    const unsigned short* pv = Vt + (j * 16 + l16) * 64 + c2 * 32 + half * 8;
                    bf.u4[0] = *(const uint4*)(pv);
                    bf.u4[1] = *(const uint4*)(pv + 16);
                    oacc[j] = __builtin_amdgcn_wmma_f32_16x16x32_bf16(
                        false, af.v, false, bf.v, (short)0, oacc[j], false, false);
                }
            }
        }
        __syncthreads();
    }

    // normalize and write [S,B,NH*HD] bf16
#pragma unroll
    for (int j = 0; j < 8; j++) {
#pragma unroll
        for (int v = 0; v < 8; v++) {
            int s = q0 + srow + v;
            float val = oacc[j][v] / li[v];
            Ob[((size_t)(s * BATCH + b) * NH + h) * HD + j * 16 + l16] =
                f32_to_bf16(val);
        }
    }
}

// ---------------------------------------------------------------------------
// Launch: cast -> QKV GEMMs -> RoPE/repack -> flash attention -> output GEMM
// ---------------------------------------------------------------------------
extern "C" void kernel_launch(void* const* d_in, const int* in_sizes, int n_in,
                              void* d_out, int out_size, void* d_ws, size_t ws_size,
                              hipStream_t stream) {
    (void)in_sizes; (void)n_in; (void)out_size; (void)ws_size;
    const float* x  = (const float*)d_in[0];
    const float* Wq = (const float*)d_in[1];
    const float* Wk = (const float*)d_in[2];
    const float* Wv = (const float*)d_in[3];
    const float* Wo = (const float*)d_in[4];
    float* out = (float*)d_out;
    char* ws = (char*)d_ws;

    const size_t MB = 1024u * 1024u;
    // live ranges are disjoint -> alias to keep scratch at 84 MB
    unsigned short* xb  = (unsigned short*)(ws + 0);        // 16MB
    unsigned short* Wqb = (unsigned short*)(ws + 16 * MB);  // 8MB
    unsigned short* Wkb = (unsigned short*)(ws + 24 * MB);  // 2MB
    unsigned short* Wvb = (unsigned short*)(ws + 26 * MB);  // 2MB
    unsigned short* Wob = (unsigned short*)(ws + 28 * MB);  // 8MB
    float* Qf = (float*)(ws + 36 * MB);                     // 32MB
    float* Kf = (float*)(ws + 68 * MB);                     // 8MB
    float* Vf = (float*)(ws + 76 * MB);                     // 8MB
    unsigned short* Qb  = (unsigned short*)(ws + 0);        // aliases xb (dead)
    unsigned short* Kb  = (unsigned short*)(ws + 16 * MB);  // aliases Wqb (dead)
    unsigned short* Vb  = (unsigned short*)(ws + 20 * MB);
    unsigned short* AOb = (unsigned short*)(ws + 36 * MB);  // aliases Qf (dead)

    const int R = S_LEN * BATCH;                            // 4096 rows

    cast_bf16_kernel<<<4096, 256, 0, stream>>>(x,  xb,  R * DMODEL);
    cast_bf16_kernel<<<2048, 256, 0, stream>>>(Wq, Wqb, DMODEL * NH * HD);
    cast_bf16_kernel<<<512,  256, 0, stream>>>(Wk, Wkb, DMODEL * NKV * HD);
    cast_bf16_kernel<<<512,  256, 0, stream>>>(Wv, Wvb, DMODEL * NKV * HD);
    cast_bf16_kernel<<<2048, 256, 0, stream>>>(Wo, Wob, NH * HD * DMODEL);

    gemm_bf16_wmma<<<dim3(2048 / 128, R / 128), 256, 0, stream>>>(xb, Wqb, Qf, R, 2048, DMODEL);
    gemm_bf16_wmma<<<dim3(512 / 128,  R / 128), 256, 0, stream>>>(xb, Wkb, Kf, R, 512,  DMODEL);
    gemm_bf16_wmma<<<dim3(512 / 128,  R / 128), 256, 0, stream>>>(xb, Wvb, Vf, R, 512,  DMODEL);

    const float scale = 0.0883883476483184f;                // 1/sqrt(128)
    int nq = S_LEN * BATCH * NH * (HD / 2);
    rope_pack_kernel<<<(nq + 255) / 256, 256, 0, stream>>>(Qf, Qb, NH, scale);
    int nk = S_LEN * BATCH * NKV * (HD / 2);
    rope_pack_kernel<<<(nk + 255) / 256, 256, 0, stream>>>(Kf, Kb, NKV, 1.0f);
    int nv = S_LEN * BATCH * NKV * HD;
    pack_v_kernel<<<(nv + 255) / 256, 256, 0, stream>>>(Vf, Vb);

    flash_attn_kernel<<<dim3(S_LEN / 128, NH, BATCH), 256, 0, stream>>>(Qb, Kb, Vb, AOb);

    gemm_bf16_wmma<<<dim3(2048 / 128, R / 128), 256, 0, stream>>>(AOb, Wob, out, R, 2048, DMODEL);
}
